// SelfAttention_69320772157860
// MI455X (gfx1250) — compile-verified, matrix operand order checked by reading, not run on hardware
//
#include <hip/hip_runtime.h>

// ---------------------------------------------------------------------------
// Problem constants (from reference): B=4, S=2048, C=512, H=8, DK=DV=64, KS=3
// ---------------------------------------------------------------------------
constexpr int BB   = 4;
constexpr int SEQ  = 2048;
constexpr int CH   = 512;
constexpr int HDS  = 8;
constexpr int DKV  = 64;          // DK == DV == 64
constexpr int NOUT = HDS * DKV;   // 512

typedef _Float16 f16;
typedef __attribute__((ext_vector_type(16))) _Float16 v16h;
typedef __attribute__((ext_vector_type(8)))  _Float16 v8h;
typedef __attribute__((ext_vector_type(8)))  float    v8f;
typedef __attribute__((ext_vector_type(4)))  float    v4f;

union Frag {
    v16h v;
    v8h  h[2];
};

// D = A(16x32 f16) * B(32x16 f16) + C(16x16 f32), wave32 WMMA.
__device__ __forceinline__ v8f wmma_f16(v16h a, v16h b, v8f c) {
    return __builtin_amdgcn_wmma_f32_16x16x32_f16(
        /*neg_a=*/false, a, /*neg_b=*/false, b,
        /*c_mod=*/(short)0, c, /*reuse_a=*/false, /*reuse_b=*/false);
}

// Load a 16x32 operand fragment from a row-major f16 matrix.
// ISA 16-bit A layout (7.12.2): lane l<16 holds row M=l, K = {k0..k0+7} U {k0+16..k0+23}
// with k0 = kbase; lane l+16 holds the same row with k0 = kbase+8.
// The B operand is loaded as rows of B^T with the same striping.
__device__ __forceinline__ v16h load_frag(const f16* __restrict__ base,
                                          int row_stride, int r, int kofs) {
    Frag f;
    const f16* p = base + (size_t)r * row_stride + kofs;
    f.h[0] = *(const v8h*)p;
    f.h[1] = *(const v8h*)(p + 16);
    return f.v;
}

// ---------------------------------------------------------------------------
// Pre-pass: f32 -> f16 conversion and conv-weight repack
// ---------------------------------------------------------------------------
__global__ void cvt_f32_f16_kernel(const float* __restrict__ in,
                                   f16* __restrict__ out, int n) {
    int i = blockIdx.x * 256 + threadIdx.x;
    if (i < n) out[i] = (f16)in[i];
}

// in: [O=512][C=512][KS=3] f32 -> out: [KS=3][O=512][C=512] f16
__global__ void repack_convw_kernel(const float* __restrict__ in,
                                    f16* __restrict__ out) {
    int i = blockIdx.x * 256 + threadIdx.x;
    if (i < 3 * CH * CH) {
        int j   = i / (CH * CH);
        int rem = i - j * (CH * CH);
        int o   = rem >> 9;
        int c   = rem & 511;
        out[i] = (f16)in[((size_t)o * CH + c) * 3 + j];
    }
}

// ---------------------------------------------------------------------------
// QKV projection: q = conv3(x)@Wq^T * 0.125, k = conv3(x)@Wk^T, v = x@Wv^T
// q,k stored [B*H][S][64] f16; v stored transposed [B*H][64][S] f16.
// Block: 4 waves (128 thr); each wave = 16 rows x 64 cols of the GEMM.
// ---------------------------------------------------------------------------
__global__ __launch_bounds__(128)
void proj_kernel(const f16* __restrict__ xh,
                 const f16* __restrict__ wqh, const f16* __restrict__ wkh,
                 const f16* __restrict__ wvh,
                 const float* __restrict__ bq, const float* __restrict__ bk,
                 const float* __restrict__ bv,
                 f16* __restrict__ qh, f16* __restrict__ kh,
                 f16* __restrict__ vT) {
    const int z   = blockIdx.z;            // 0=q, 1=k, 2=v
    const int ct0 = blockIdx.y * 64;       // output-column tile base
    const int tid = threadIdx.x;
    const int w   = tid >> 5;
    const int lane = tid & 31;
    const int r   = lane & 15;
    const int hi  = lane >> 4;

    const int m0 = blockIdx.x * 64 + w * 16;   // global row (b*S + s)
    const int b  = m0 >> 11;                   // /2048  (S divides 64-row tiles)
    const int s0 = m0 & (SEQ - 1);

    const f16*  wsel = (z == 0) ? wqh : (z == 1) ? wkh : wvh;
    const float* bias = (z == 0) ? bq : (z == 1) ? bk : bv;
    const int ntaps = (z == 2) ? 1 : 3;

    v8f acc[4];
    #pragma unroll
    for (int t = 0; t < 4; ++t) acc[t] = (v8f){};

    for (int j = 0; j < ntaps; ++j) {
        const int shift = (z == 2) ? 0 : (j - 2);  // causal left pad
        const f16* wj = wsel + (size_t)j * CH * CH;
        const int srow = s0 + r + shift;
        const bool valid = (srow >= 0);
        const f16* xrow = xh + ((size_t)b * SEQ + (valid ? srow : 0)) * CH;

        for (int kc = 0; kc < CH; kc += 32) {
            Frag fa;
            if (valid) {
                const f16* p = xrow + kc + hi * 8;
                fa.h[0] = *(const v8h*)p;
                fa.h[1] = *(const v8h*)(p + 16);
            } else {
                v16h zz = {};
                fa.v = zz;
            }
            #pragma unroll
            for (int t = 0; t < 4; ++t) {
                const int n = ct0 + t * 16 + r;
                Frag fb;
                const f16* wrow = wj + (size_t)n * CH + kc + hi * 8;
                fb.h[0] = *(const v8h*)wrow;
                fb.h[1] = *(const v8h*)(wrow + 16);
                acc[t] = wmma_f16(fa.v, fb.v, acc[t]);
            }
        }
    }

    #pragma unroll
    for (int t = 0; t < 4; ++t) {
        const int n  = ct0 + t * 16 + r;       // output column (h*64+d)
        const float bi = bias[n];
        const int hsel = n >> 6;
        const int d    = n & 63;
        #pragma unroll
        for (int v = 0; v < 8; ++v) {
            const int s = s0 + v + hi * 8;
            float val = acc[t][v] + bi;
            if (z == 0) {
                qh[(((size_t)b * HDS + hsel) * SEQ + s) * DKV + d] =
                    (f16)(val * 0.125f);       // fold 1/sqrt(DK) into Q
            } else if (z == 1) {
                kh[(((size_t)b * HDS + hsel) * SEQ + s) * DKV + d] = (f16)val;
            } else {
                vT[(((size_t)b * HDS + hsel) * DKV + d) * SEQ + s] = (f16)val;
            }
        }
    }
}

// ---------------------------------------------------------------------------
// Attention: one block = 16 query rows of one (b,h).
// Pass 1: online softmax stats over causal tiles (WMMA QK^T).
// Pass 2 (fused): per 128-col group, recompute P, stage f32 P in LDS and
//   drain it with fully coalesced 16B stores (the dominant traffic), stash
//   f16 P in LDS, and WMMA against V^T into the output accumulator.
// ---------------------------------------------------------------------------
__global__ __launch_bounds__(128)
void attn_kernel(const f16* __restrict__ qh, const f16* __restrict__ kh,
                 const f16* __restrict__ vT,
                 float* __restrict__ outp, float* __restrict__ attn) {
    const int qb  = blockIdx.x;            // 16-row query block
    const int bh  = blockIdx.y;            // b*H + h
    const int b   = bh >> 3;
    const int h   = bh & 7;
    const int tid = threadIdx.x;
    const int w   = tid >> 5;
    const int lane = tid & 31;
    const int r   = lane & 15;
    const int hi  = lane >> 4;

    __shared__ float red_m[4][16];
    __shared__ float red_s[4][16];
    __shared__ float Mrow[16];
    __shared__ float SinvRow[16];
    __shared__ f16   Plds[16][136];        // f16 P for PV A-frags (pad: no conflicts)
    __shared__ float Plds32[16][132];      // f32 P staged for coalesced stores

    const f16* qbase = qh + ((size_t)bh * SEQ + (size_t)qb * 16) * DKV;
    const f16* kbase = kh + (size_t)bh * SEQ * DKV;
    const f16* vbase = vT + (size_t)bh * DKV * SEQ;
    float* attnBase = attn + ((size_t)bh * SEQ + (size_t)qb * 16) * SEQ;

    // Q fragments (kept live for both passes); Q already has the 1/8 scale.
    const v16h aq0 = load_frag(qbase, DKV, r, 0 + hi * 8);
    const v16h aq1 = load_frag(qbase, DKV, r, 32 + hi * 8);

    // ---- Pass 1: online row max / exp-sum over this wave's column tiles ----
    float m[8], sum[8];
    #pragma unroll
    for (int v = 0; v < 8; ++v) { m[v] = -3.0e38f; sum[v] = 0.0f; }

    for (int ct = w; ct <= qb; ct += 4) {
        if (ct + 4 <= qb)
            __builtin_prefetch(kbase + (size_t)(ct + 4) * 16 * DKV, 0, 1);
        const f16* ktile = kbase + (size_t)ct * 16 * DKV;
        v16h b0 = load_frag(ktile, DKV, r, 0 + hi * 8);
        v16h b1 = load_frag(ktile, DKV, r, 32 + hi * 8);
        v8f e = {};
        e = wmma_f16(aq0, b0, e);
        e = wmma_f16(aq1, b1, e);
        const int col = ct * 16 + r;
        #pragma unroll
        for (int v = 0; v < 8; ++v) {
            const int qrow = qb * 16 + v + hi * 8;
            float ev = (col > qrow) ? -3.0e38f : e[v];
            float tmax = ev;
            tmax = fmaxf(tmax, __shfl_xor(tmax, 1, 16));
            tmax = fmaxf(tmax, __shfl_xor(tmax, 2, 16));
            tmax = fmaxf(tmax, __shfl_xor(tmax, 4, 16));
            tmax = fmaxf(tmax, __shfl_xor(tmax, 8, 16));
            const float nm = fmaxf(m[v], tmax);
            float pe = __expf(ev - nm);
            pe += __shfl_xor(pe, 1, 16);
            pe += __shfl_xor(pe, 2, 16);
            pe += __shfl_xor(pe, 4, 16);
            pe += __shfl_xor(pe, 8, 16);
            sum[v] = sum[v] * __expf(m[v] - nm) + pe;
            m[v] = nm;
        }
    }

    if (lane == 0) {
        #pragma unroll
        for (int v = 0; v < 8; ++v) { red_m[w][v] = m[v]; red_s[w][v] = sum[v]; }
    }
    if (lane == 16) {
        #pragma unroll
        for (int v = 0; v < 8; ++v) { red_m[w][8 + v] = m[v]; red_s[w][8 + v] = sum[v]; }
    }
    __syncthreads();
    if (tid < 16) {
        float M = red_m[0][tid];
        M = fmaxf(M, red_m[1][tid]);
        M = fmaxf(M, red_m[2][tid]);
        M = fmaxf(M, red_m[3][tid]);
        float Ssum = 0.0f;
        #pragma unroll
        for (int w2 = 0; w2 < 4; ++w2)
            Ssum += red_s[w2][tid] * __expf(red_m[w2][tid] - M);
        Mrow[tid]    = M;
        SinvRow[tid] = 1.0f / Ssum;        // >0: the diagonal col is always live
    }
    __syncthreads();

    const int ngroups = (qb + 8) >> 3;     // ceil((qb+1)*16 / 128)

    // ---- Zero-fill attention columns beyond the group-covered region -------
    {
        const int c0 = ngroups * 128;      // groups write [0, c0)
        const int rq = (SEQ - c0) >> 2;    // float4 per row
        if (rq > 0) {
            const v4f z4 = {};
            for (int i = tid; i < 16 * rq; i += 128) {
                const int row = i / rq;
                const int cq  = i - row * rq;
                *(v4f*)(attnBase + (size_t)row * SEQ + c0 + cq * 4) = z4;
            }
        }
    }

    // ---- Pass 2 fused: P stage + coalesced store + P@V, 128 cols/group -----
    v8f acc = {};
    for (int g = 0; g < ngroups; ++g) {
        #pragma unroll
        for (int i = 0; i < 2; ++i) {
            const int ct  = g * 8 + 2 * w + i;   // global 16-col tile
            const int ctg = 2 * w + i;           // tile within group
            if (ct <= qb) {
                const f16* ktile = kbase + (size_t)ct * 16 * DKV;
                v16h b0 = load_frag(ktile, DKV, r, 0 + hi * 8);
                v16h b1 = load_frag(ktile, DKV, r, 32 + hi * 8);
                v8f e = {};
                e = wmma_f16(aq0, b0, e);
                e = wmma_f16(aq1, b1, e);
                const int col = ct * 16 + r;
                #pragma unroll
                for (int v = 0; v < 8; ++v) {
                    const int row  = v + hi * 8;
                    const int qrow = qb * 16 + row;
                    float ev = (col > qrow) ? -3.0e38f : e[v];
                    float p = __expf(ev - Mrow[row]) * SinvRow[row];
                    Plds32[row][ctg * 16 + r] = p;
                    Plds[row][ctg * 16 + r]   = (f16)p;
                }
            } else {
                #pragma unroll
                for (int v = 0; v < 8; ++v) {
                    const int row = v + hi * 8;
                    Plds32[row][ctg * 16 + r] = 0.0f;
                    Plds[row][ctg * 16 + r]   = (f16)0.0f;
                }
            }
        }
        __syncthreads();

        // Drain the 16x128 f32 P tile: one 512B contiguous burst per wave op.
        #pragma unroll
        for (int it = 0; it < 4; ++it) {
            const int q   = tid + it * 128;      // 0..511 quads
            const int row = q >> 5;
            const int cq  = q & 31;
            v4f val = *(const v4f*)&Plds32[row][cq * 4];
            *(v4f*)(attnBase + (size_t)row * SEQ + g * 128 + cq * 4) = val;
        }

        // P(16x128) @ Vtile(128x16): 4 WMMA K-steps; this wave owns 16 out cols.
        #pragma unroll
        for (int c4 = 0; c4 < 4; ++c4) {
            Frag fa;
            const f16* prow = &Plds[r][c4 * 32 + hi * 8];
            fa.h[0] = *(const v8h*)prow;
            fa.h[1] = *(const v8h*)(prow + 16);
            Frag fb;
            const int n = w * 16 + r;                  // out col within head
            const f16* vrow = vbase + (size_t)n * SEQ + g * 128 + c4 * 32 + hi * 8;
            fb.h[0] = *(const v8h*)vrow;
            fb.h[1] = *(const v8h*)(vrow + 16);
            acc = wmma_f16(fa.v, fb.v, acc);
        }
        __syncthreads();
    }

    // ---- Output: out[b, s, h*64 + wave-col] -------------------------------
    const int ocol = h * DKV + w * 16 + r;
    #pragma unroll
    for (int v = 0; v < 8; ++v) {
        const int s = qb * 16 + v + hi * 8;
        outp[((size_t)b * SEQ + s) * NOUT + ocol] = acc[v];
    }
}

// ---------------------------------------------------------------------------
// Host launcher
// ---------------------------------------------------------------------------
extern "C" void kernel_launch(void* const* d_in, const int* in_sizes, int n_in,
                              void* d_out, int out_size, void* d_ws, size_t ws_size,
                              hipStream_t stream) {
    (void)in_sizes; (void)n_in; (void)out_size; (void)ws_size;

    const float* x  = (const float*)d_in[0];
    // d_in[1] = mask (implicit causal; unused)
    const float* wq = (const float*)d_in[2];
    const float* bq = (const float*)d_in[3];
    const float* wk = (const float*)d_in[4];
    const float* bk = (const float*)d_in[5];
    const float* wv = (const float*)d_in[6];
    const float* bv = (const float*)d_in[7];

    float* outp = (float*)d_out;
    float* attn = outp + (size_t)BB * SEQ * NOUT;    // tuple: (out, attention)

    // Workspace carve-up (~35.5 MB total)
    char* ws = (char*)d_ws;
    f16* xh  = (f16*)(ws);                            // 8,388,608 B
    f16* wqh = (f16*)(ws + 8388608);                  // 1,572,864 B
    f16* wkh = (f16*)(ws + 9961472);                  // 1,572,864 B
    f16* wvh = (f16*)(ws + 11534336);                 //   524,288 B
    f16* qh  = (f16*)(ws + 12058624);                 // 8,388,608 B
    f16* kh  = (f16*)(ws + 20447232);                 // 8,388,608 B
    f16* vT  = (f16*)(ws + 28835840);                 // 8,388,608 B

    const int nx = BB * SEQ * CH;                     // 4,194,304
    cvt_f32_f16_kernel<<<nx / 256, 256, 0, stream>>>(x, xh, nx);
    repack_convw_kernel<<<(3 * CH * CH) / 256, 256, 0, stream>>>(wq, wqh);
    repack_convw_kernel<<<(3 * CH * CH) / 256, 256, 0, stream>>>(wk, wkh);
    cvt_f32_f16_kernel<<<(CH * CH) / 256, 256, 0, stream>>>(wv, wvh, CH * CH);

    proj_kernel<<<dim3((BB * SEQ) / 64, NOUT / 64, 3), 128, 0, stream>>>(
        xh, wqh, wkh, wvh, bq, bk, bv, qh, kh, vT);

    attn_kernel<<<dim3(SEQ / 16, BB * HDS), 128, 0, stream>>>(
        qh, kh, vT, outp, attn);
}
